// SortingModel_29446295781546
// MI455X (gfx1250) — compile-verified, matrix operand order checked by reading, not run on hardware
//
#include <hip/hip_runtime.h>
#include <hip/hip_bf16.h>
#include <cstdint>

// ---------------------------------------------------------------------------
// Problem constants (from reference)
// ---------------------------------------------------------------------------
#define BB     256          // batch
#define NN     200          // permutation size
#define NU     32           // hidden units (== WMMA K!)
#define SS     10           // samples per num
#define NITER  20           // sinkhorn iterations
#define LDSTR  201          // padded LDS row stride (odd -> conflict-free cols)

typedef __attribute__((ext_vector_type(16))) _Float16 v16h;
typedef __attribute__((ext_vector_type(8)))  float    v8f;

#ifndef __has_builtin
#define __has_builtin(x) 0
#endif

#if __has_builtin(__builtin_amdgcn_global_load_async_to_lds_b32)
#define ASYNC_LDS 1
#else
#define ASYNC_LDS 0
#endif

#define AS1 __attribute__((address_space(1)))
#define AS3 __attribute__((address_space(3)))

// ---------------------------------------------------------------------------
// Kernel 0: zero the scalar output (accumulated via atomics later)
// ---------------------------------------------------------------------------
__global__ void zero_out_kernel(float* __restrict__ out) {
    if (threadIdx.x == 0 && blockIdx.x == 0) out[0] = 0.0f;
}

// ---------------------------------------------------------------------------
// Kernel 1: MLP -> log_alpha via WMMA.
//   log_alpha[m][n] = sum_k relu(x[m]*W1[k]+b1[k]) * W2[n][k] + b2[n]
//   M = B*N = 51200 rows, N = 200 cols, K = 32  (one v_wmma_f32_16x16x32_f16
//   per 16x16 output tile; A/B built per-lane following ISA 7.12.2 layouts)
// One wave per 16x16 tile; 8 waves / block; grid = (51200/16/8, ceil(200/16)).
// ---------------------------------------------------------------------------
__device__ __forceinline__ void load8(float* dst, const float* src) {
    *(float4*)(dst + 0) = *(const float4*)(src + 0);
    *(float4*)(dst + 4) = *(const float4*)(src + 4);
}

__global__ __launch_bounds__(256) void mlp_wmma_kernel(
    const float* __restrict__ random,   // (B*N) flat
    const float* __restrict__ W1,       // (32)
    const float* __restrict__ b1,       // (32)
    const float* __restrict__ W2,       // (200,32) row-major
    const float* __restrict__ b2,       // (200)
    float* __restrict__ log_alpha)      // (B*N, 200) workspace
{
    const int lane   = threadIdx.x & 31;
    const int wv     = threadIdx.x >> 5;
    const int tileM  = blockIdx.x * 8 + wv;      // 0..3199
    const int tileN  = blockIdx.y;               // 0..12
    const int half16 = lane >> 4;                // 0 | 1
    const int l16    = lane & 15;
    const int klo    = half16 * 8;               // K base per lane-half
    // per-lane K set: element i<8 -> k = klo+i ; i>=8 -> k = 16+klo+(i-8)

    // ---- branch-free vector loads of the K slices --------------------------
    float w1v[16], b1v[16], w2v[16];
    load8(w1v + 0, W1 + klo);
    load8(w1v + 8, W1 + 16 + klo);
    load8(b1v + 0, b1 + klo);
    load8(b1v + 8, b1 + 16 + klo);

    const int   n     = tileN * 16 + l16;
    const bool  nok   = (n < NN);
    const int   n_c   = nok ? n : (NN - 1);      // clamp row, zero later
    const float nmask = nok ? 1.0f : 0.0f;
    load8(w2v + 0, W2 + n_c * NU + klo);
    load8(w2v + 8, W2 + n_c * NU + 16 + klo);

    // ---- A matrix (16x32 f16): h[m][k] = relu(x[m]*W1[k]+b1[k]) ------------
    const int   m  = tileM * 16 + l16;           // always < 51200
    const float xm = random[m];
    v16h a;
#pragma unroll
    for (int i = 0; i < 16; ++i)
        a[i] = (_Float16)fmaxf(xm * w1v[i] + b1v[i], 0.0f);

    // ---- B matrix (32x16 f16): B[k][n] = W2[n][k] --------------------------
    v16h bm;
#pragma unroll
    for (int i = 0; i < 16; ++i)
        bm[i] = (_Float16)(w2v[i] * nmask);

    // ---- C (16x16 f32): preload bias b2[n] ---------------------------------
    v8f c;
    const float cb = nok ? b2[n] : 0.0f;
#pragma unroll
    for (int i = 0; i < 8; ++i) c[i] = cb;

    // D = A x B + C   (emits v_wmma_f32_16x16x32_f16)
    v8f d = __builtin_amdgcn_wmma_f32_16x16x32_f16(
        /*neg_a=*/false, a, /*neg_b=*/false, bm,
        /*c_mod=*/(short)0, c, /*reuse_a=*/false, /*reuse_b=*/false);

    // ---- store: D VGPR r holds (M=r | r+8 by lane-half), N = l16 -----------
    if (nok) {
#pragma unroll
        for (int r = 0; r < 8; ++r) {
            const int row = tileM * 16 + r + 8 * half16;
            log_alpha[(size_t)row * NN + n] = d[r];
        }
    }
}

// ---------------------------------------------------------------------------
// Kernel 2: Gumbel-Sinkhorn + loss. One workgroup (1024 thr = 32 waves) per
// (sample, batch) matrix; the 200x200 matrix lives in LDS (200*201*4 =
// 160,800 B -> two workgroups fit in the 320 KB WGP LDS). Exp-bound core:
// wave-per-row / wave-per-column with wave32 shuffle reductions.
// ---------------------------------------------------------------------------
__global__ __launch_bounds__(1024, 1) void sinkhorn_kernel(
    const float* __restrict__ log_alpha,     // (B*N, N) from kernel 1
    const float* __restrict__ noise,         // (S*B, N, N)
    const float* __restrict__ random,        // (B, N)
    const float* __restrict__ ordered_tiled, // (S*B, N)
    float* __restrict__ out)                 // scalar accumulator
{
    extern __shared__ float la[];            // [NN][LDSTR]
    __shared__ float blk_sum;

    const int idx  = blockIdx.x;             // 0 .. S*B-1 (sample-major)
    const int b    = idx % BB;
    const int tid  = threadIdx.x;
    const int lane = tid & 31;
    const int wv   = tid >> 5;               // 0..31
    const int nthr = blockDim.x;

    const float* la_g = log_alpha + (size_t)b   * (NN * NN);
    const float* nz_g = noise     + (size_t)idx * (NN * NN);
    const float* rnd  = random        + (size_t)b   * NN;
    const float* ord  = ordered_tiled + (size_t)idx * NN;

    // warm L2 for the tail reads (global_prefetch_b8)
    if (tid < NN) {
        __builtin_prefetch(rnd + tid, 0, 1);
        __builtin_prefetch(ord + tid, 0, 1);
    }

#if ASYNC_LDS
    // ---- fill LDS with the gumbel noise via async copy (ASYNCcnt path) -----
    for (int g = tid; g < NN * NN; g += nthr) {
        const int r  = g / NN;               // const divisor -> mul/shift
        const int cc = g - r * NN;
        __builtin_amdgcn_global_load_async_to_lds_b32(
            (AS1 int*)(uintptr_t)(nz_g + g),
            (AS3 int*)(uintptr_t)(la + r * LDSTR + cc),
            /*offset=*/0, /*cpol=*/0);
    }
#if __has_builtin(__builtin_amdgcn_s_wait_asynccnt)
    __builtin_amdgcn_s_wait_asynccnt(0);
#else
    asm volatile("s_wait_asynccnt 0x0" ::: "memory");
#endif
    __syncthreads();
    // add the (sample-shared) log_alpha on top
    for (int g = tid; g < NN * NN; g += nthr) {
        const int r  = g / NN;
        const int cc = g - r * NN;
        la[r * LDSTR + cc] += la_g[g];
    }
#else
    // ---- fallback: fused load la = log_alpha[b] + noise[idx] ---------------
    for (int g = tid; g < NN * NN; g += nthr) {
        const int r  = g / NN;
        const int cc = g - r * NN;
        la[r * LDSTR + cc] = la_g[g] + nz_g[g];
    }
#endif
    if (tid == 0) blk_sum = 0.0f;
    __syncthreads();

    // ---- 20 Sinkhorn iterations --------------------------------------------
    for (int it = 0; it < NITER; ++it) {
        // row normalize (logsumexp over columns), one wave per row
        for (int r = wv; r < NN; r += 32) {
            float* row = la + r * LDSTR;
            float mx = -__builtin_inff();
            for (int cc = lane; cc < NN; cc += 32) mx = fmaxf(mx, row[cc]);
#pragma unroll
            for (int o = 16; o > 0; o >>= 1) mx = fmaxf(mx, __shfl_xor(mx, o, 32));
            float s = 0.0f;
            for (int cc = lane; cc < NN; cc += 32) s += __expf(row[cc] - mx);
#pragma unroll
            for (int o = 16; o > 0; o >>= 1) s += __shfl_xor(s, o, 32);
            const float lse = mx + __logf(s);
            for (int cc = lane; cc < NN; cc += 32) row[cc] -= lse;
        }
        __syncthreads();
        // column normalize (logsumexp over rows), one wave per column
        // (stride LDSTR=201 is odd -> lanes hit distinct LDS banks)
        for (int cc = wv; cc < NN; cc += 32) {
            float mx = -__builtin_inff();
            for (int r = lane; r < NN; r += 32) mx = fmaxf(mx, la[r * LDSTR + cc]);
#pragma unroll
            for (int o = 16; o > 0; o >>= 1) mx = fmaxf(mx, __shfl_xor(mx, o, 32));
            float s = 0.0f;
            for (int r = lane; r < NN; r += 32) s += __expf(la[r * LDSTR + cc] - mx);
#pragma unroll
            for (int o = 16; o > 0; o >>= 1) s += __shfl_xor(s, o, 32);
            const float lse = mx + __logf(s);
            for (int r = lane; r < NN; r += 32) la[r * LDSTR + cc] -= lse;
        }
        __syncthreads();
    }

    // ---- rec_i = sum_j exp(la[j][i]) * random[b][j]; loss = MSE ------------
    float wave_part = 0.0f;
    for (int i = wv; i < NN; i += 32) {
        float rec = 0.0f;
        for (int j = lane; j < NN; j += 32)
            rec += __expf(la[j * LDSTR + i]) * rnd[j];
#pragma unroll
        for (int o = 16; o > 0; o >>= 1) rec += __shfl_xor(rec, o, 32);
        if (lane == 0) {
            const float diff = ord[i] - rec;
            wave_part += diff * diff;
        }
    }
    if (lane == 0) atomicAdd(&blk_sum, wave_part);
    __syncthreads();
    if (tid == 0)
        atomicAdd(out, blk_sum * (1.0f / ((float)SS * (float)BB * (float)NN)));
}

// ---------------------------------------------------------------------------
// Host-side launcher
// ---------------------------------------------------------------------------
extern "C" void kernel_launch(void* const* d_in, const int* in_sizes, int n_in,
                              void* d_out, int out_size, void* d_ws, size_t ws_size,
                              hipStream_t stream) {
    const float* random        = (const float*)d_in[0];  // (B, N)
    const float* ordered_tiled = (const float*)d_in[1];  // (S*B, N)
    // d_in[2] random_tiled == tile(random) -> recomputed via b = idx % B
    const float* noise         = (const float*)d_in[3];  // (S*B, N, N)
    const float* W1            = (const float*)d_in[4];  // (32, 1)
    const float* b1            = (const float*)d_in[5];  // (32)
    const float* W2            = (const float*)d_in[6];  // (200, 32)
    const float* b2            = (const float*)d_in[7];  // (200)
    float* out       = (float*)d_out;
    float* log_alpha = (float*)d_ws;                     // (B*N, 200) = 40.96 MB

    zero_out_kernel<<<1, 64, 0, stream>>>(out);

    // 51200 rows / 16 = 3200 M-tiles; 8 waves/block -> 400 blocks; 13 N-tiles
    mlp_wmma_kernel<<<dim3(400, 13), 256, 0, stream>>>(
        random, W1, b1, W2, b2, log_alpha);

    const size_t lds_bytes = (size_t)NN * LDSTR * sizeof(float); // 160,800 B
    sinkhorn_kernel<<<SS * BB, 1024, lds_bytes, stream>>>(
        log_alpha, noise, random, ordered_tiled, out);
}
